// DeepQNetwork_80917183856871
// MI455X (gfx1250) — matrix operand order repaired and checked by reading, not compiled, not run on hardware
//
#include <hip/hip_runtime.h>

typedef __bf16 bf16;
typedef __attribute__((ext_vector_type(16))) __bf16 bf16x16;
typedef __attribute__((ext_vector_type(8)))  __bf16 bf16x8;
typedef __attribute__((ext_vector_type(8)))  float  v8f;
typedef __attribute__((ext_vector_type(4)))  unsigned int uint32x4;
typedef __attribute__((ext_vector_type(8)))  int int32x8;
typedef __attribute__((ext_vector_type(4)))  int int32x4;

#define DEVINL __device__ __forceinline__

// ---- WMMA fragment helpers (CDNA5 wave32, 16x16x32 bf16) ----------------
// A (16x32, row-major, leading-dim ldk): lane L: m=L&15, half=L>>4
//   elems 0..7 = K[8*half .. +7], elems 8..15 = K[16+8*half .. +7]
DEVINL bf16x16 frag_a(const bf16* row /* = base + m*ldk + kk */, int half) {
    bf16x8 lo = *(const bf16x8*)(row + 8 * half);
    bf16x8 hi = *(const bf16x8*)(row + 16 + 8 * half);
    return __builtin_shufflevector(lo, hi, 0,1,2,3,4,5,6,7,8,9,10,11,12,13,14,15);
}
// B (32x16, stored transposed [n][k] row-major): lane L: n=L&15, half=L>>4,
//   elems 0..7 = K[8*half .. +7], elems 8..15 = K[16+8*half .. +7]
DEVINL bf16x16 frag_b(const bf16* row /* = base + n*ldk + kk */, int half) {
    bf16x8 lo = *(const bf16x8*)(row + 8 * half);
    bf16x8 hi = *(const bf16x8*)(row + 16 + 8 * half);
    return __builtin_shufflevector(lo, hi, 0,1,2,3,4,5,6,7,8,9,10,11,12,13,14,15);
}
DEVINL v8f wmma_bf16(bf16x16 a, bf16x16 b, v8f c) {
    return __builtin_amdgcn_wmma_f32_16x16x32_bf16(false, a, false, b, (short)0, c,
                                                   false, false);
}
DEVINL float relu(float v) { return v > 0.f ? v : 0.f; }

// ---- TDM: 2D tile load global->LDS (D# per CDNA5 ISA ch.8) ---------------
// tile: rows x tileElems (bf16, data_size=2B), row stride = strideElems.
// LDS dest: rows packed contiguously (tileElems*2 bytes per row).
DEVINL void tdm_load_2d(unsigned int ldsOff, const void* gptr,
                        unsigned int tileElems, unsigned int rows,
                        unsigned int strideElems,
                        unsigned int tensorDim0, unsigned int tensorDim1)
{
    unsigned long long ga = (unsigned long long)(size_t)gptr;
    uint32x4 g0;
    g0[0] = 1u;                                   // count=1, user descriptor
    g0[1] = ldsOff;                               // lds_addr (bytes)
    g0[2] = (unsigned int)ga;                     // global_addr[31:0]
    g0[3] = (unsigned int)(ga >> 32) | (2u << 30);// global_addr[56:32] | type=2
    int32x8 g1;
    g1[0] = (int)(1u << 16);                      // wg_mask=0, data_size=1 (2B)
    g1[1] = (int)((tensorDim0 & 0xFFFFu) << 16);  // tensor_dim0[15:0]
    g1[2] = (int)((tensorDim0 >> 16) | ((tensorDim1 & 0xFFFFu) << 16));
    g1[3] = (int)((tensorDim1 >> 16) | (tileElems << 16));  // tile_dim0
    g1[4] = (int)(rows & 0xFFFFu);                // tile_dim1, tile_dim2=0
    g1[5] = (int)strideElems;                     // tensor_dim0_stride[31:0]
    g1[6] = 0;
    g1[7] = 0;
    int32x4 z4 = {0, 0, 0, 0};
    int32x8 z8 = {0, 0, 0, 0, 0, 0, 0, 0};
    __builtin_amdgcn_tensor_load_to_lds(g0, g1, z4, z4, z8, 0);
}

// ---- Kernel 0: repack all weights fp32 -> bf16, WMMA-friendly layouts ----
__global__ __launch_bounds__(256) void repack_kernel(
    const float* __restrict__ k1, const float* __restrict__ k2,
    const float* __restrict__ k3, const float* __restrict__ W1,
    const float* __restrict__ W2, const float* __restrict__ W3,
    const float* __restrict__ W4, const float* __restrict__ W5,
    bf16* __restrict__ w1p, bf16* __restrict__ w2p, bf16* __restrict__ w3p,
    bf16* __restrict__ W1p, bf16* __restrict__ W2p, bf16* __restrict__ W3p,
    bf16* __restrict__ W4p, bf16* __restrict__ W5p)
{
    const int N1 = 32 * 192, N2 = 64 * 512, N3 = 64 * 576;
    const int NM1 = 8 * 64 * 3136, NMx = 8 * 64 * 64;
    const int total = N1 + N2 + N3 + NM1 + 4 * NMx;
    for (int i = blockIdx.x * blockDim.x + threadIdx.x; i < total;
         i += gridDim.x * blockDim.x) {
        int idx = i;
        if (idx < N1) {
            int o = idx / 192, r = idx % 192;
            int kh = r / 24, r2 = r % 24, kw = r2 / 3, ci = r2 % 3;
            w1p[idx] = (bf16)k1[((o * 3 + ci) * 8 + kh) * 8 + kw];
            continue;
        }
        idx -= N1;
        if (idx < N2) {
            int o = idx / 512, r = idx % 512;
            int kh = r / 128, r2 = r % 128, kw = r2 / 32, ci = r2 % 32;
            w2p[idx] = (bf16)k2[((o * 32 + ci) * 4 + kh) * 4 + kw];
            continue;
        }
        idx -= N2;
        if (idx < N3) {
            int o = idx / 576, r = idx % 576;
            int kh = r / 192, r2 = r % 192, kw = r2 / 64, ci = r2 % 64;
            w3p[idx] = (bf16)k3[((o * 64 + ci) * 3 + kh) * 3 + kw];
            continue;
        }
        idx -= N3;
        if (idx < NM1) {
            int e = idx / (64 * 3136), r = idx % (64 * 3136);
            int n = r / 3136, kp = r % 3136;
            int p = kp / 64, c = kp % 64;
            W1p[idx] = (bf16)W1[((size_t)e * 3136 + c * 49 + p) * 64 + n];
            continue;
        }
        idx -= NM1;
        int which = idx / NMx, r = idx % NMx;
        int e = r / 4096, r2 = r % 4096, n = r2 / 64, k = r2 & 63;
        const float* Ws = (which == 0) ? W2 : (which == 1) ? W3 : (which == 2) ? W4 : W5;
        bf16* Wd = (which == 0) ? W2p : (which == 1) ? W3p : (which == 2) ? W4p : W5p;
        Wd[r] = (bf16)Ws[(e * 64 + k) * 64 + n];
    }
}

// ---- Kernel 1: conv1 8x8 s4, 3->32, fused fp32->bf16 + ReLU --------------
__global__ __launch_bounds__(64) void conv1_kernel(
    const float* __restrict__ state, const bf16* __restrict__ w1p,
    const float* __restrict__ c1, bf16* __restrict__ act1)
{
    __shared__ __align__(16) bf16 aT[16 * 192];
    const int gid = blockIdx.x, b = gid / 25, tile = gid % 25;
    const int p0 = tile * 16, tid = threadIdx.x;
    for (int s = tid; s < 128; s += 64) {
        int m = s >> 3, kh = s & 7;
        int p = p0 + m, oh = p / 20, ow = p % 20;
        int ih = oh * 4 + kh, iw0 = ow * 4;
        const float* src = state + ((size_t)(b * 84 + ih) * 84 + iw0) * 3;
        bf16* dst = aT + m * 192 + kh * 24;
        #pragma unroll
        for (int j = 0; j < 24; ++j) dst[j] = (bf16)src[j];
    }
    __syncthreads();
    const int lane = tid & 31, wv = tid >> 5;
    const int half = lane >> 4, nn = lane & 15, n0 = wv * 16;
    v8f acc = {0.f, 0.f, 0.f, 0.f, 0.f, 0.f, 0.f, 0.f};
    const bf16* wrow = w1p + (n0 + nn) * 192;
    #pragma unroll
    for (int kk = 0; kk < 192; kk += 32)
        acc = wmma_bf16(frag_a(aT + nn * 192 + kk, half), frag_b(wrow + kk, half), acc);
    const float bias = c1[n0 + nn];
    #pragma unroll
    for (int r = 0; r < 8; ++r) {
        int p = p0 + r + 8 * half;  // NHWC index is linear in position p
        act1[((size_t)b * 400 + p) * 32 + (n0 + nn)] = (bf16)relu(acc[r] + bias);
    }
}

// ---- Kernel 2: conv2 4x4 s2, 32->64 --------------------------------------
__global__ __launch_bounds__(128) void conv2_kernel(
    const bf16* __restrict__ act1, const bf16* __restrict__ w2p,
    const float* __restrict__ c2, bf16* __restrict__ act2)
{
    __shared__ __align__(16) bf16 aT[16 * 512];
    const int gid = blockIdx.x, b = gid / 6, tile = gid % 6;
    const int p0 = tile * 16, tid = threadIdx.x;
    const int valid = (81 - p0 < 16) ? (81 - p0) : 16;
    for (int c = tid; c < 1024; c += 128) {
        int seg = c >> 4, inner = c & 15;
        int m = seg >> 2, kh = seg & 3;
        int p = p0 + m; p = (p > 80) ? 80 : p;
        int oh = p / 9, ow = p % 9, ih = oh * 2 + kh;
        const bf16* src = act1 + ((size_t)(b * 20 + ih) * 20 + ow * 2) * 32 + inner * 8;
        *(uint4*)(aT + m * 512 + kh * 128 + inner * 8) = *(const uint4*)src;
    }
    __syncthreads();
    const int lane = tid & 31, wv = tid >> 5;
    const int half = lane >> 4, nn = lane & 15, n0 = wv * 16;
    v8f acc = {0.f, 0.f, 0.f, 0.f, 0.f, 0.f, 0.f, 0.f};
    const bf16* wrow = w2p + (n0 + nn) * 512;
    #pragma unroll
    for (int kk = 0; kk < 512; kk += 32)
        acc = wmma_bf16(frag_a(aT + nn * 512 + kk, half), frag_b(wrow + kk, half), acc);
    const float bias = c2[n0 + nn];
    #pragma unroll
    for (int r = 0; r < 8; ++r) {
        int m = r + 8 * half;
        if (m < valid)
            act2[((size_t)b * 81 + p0 + m) * 64 + (n0 + nn)] = (bf16)relu(acc[r] + bias);
    }
}

// ---- Kernel 3: conv3 3x3 s1, 64->64, output = NCHW-matched flat ----------
__global__ __launch_bounds__(128) void conv3_kernel(
    const bf16* __restrict__ act2, const bf16* __restrict__ w3p,
    const float* __restrict__ c3, bf16* __restrict__ act3)
{
    __shared__ __align__(16) bf16 aT[16 * 576];
    const int gid = blockIdx.x, b = gid / 4, tile = gid % 4;
    const int p0 = tile * 16, tid = threadIdx.x;
    const int valid = (49 - p0 < 16) ? (49 - p0) : 16;
    for (int c = tid; c < 1152; c += 128) {
        int seg = c / 24, inner = c % 24;
        int m = seg / 3, kh = seg % 3;
        int p = p0 + m; p = (p > 48) ? 48 : p;
        int oh = p / 7, ow = p % 7, ih = oh + kh;
        const bf16* src = act2 + ((size_t)(b * 9 + ih) * 9 + ow) * 64 + inner * 8;
        *(uint4*)(aT + m * 576 + kh * 192 + inner * 8) = *(const uint4*)src;
    }
    __syncthreads();
    const int lane = tid & 31, wv = tid >> 5;
    const int half = lane >> 4, nn = lane & 15, n0 = wv * 16;
    v8f acc = {0.f, 0.f, 0.f, 0.f, 0.f, 0.f, 0.f, 0.f};
    const bf16* wrow = w3p + (n0 + nn) * 576;
    #pragma unroll
    for (int kk = 0; kk < 576; kk += 32)
        acc = wmma_bf16(frag_a(aT + nn * 576 + kk, half), frag_b(wrow + kk, half), acc);
    const float bias = c3[n0 + nn];
    #pragma unroll
    for (int r = 0; r < 8; ++r) {
        int m = r + 8 * half;
        if (m < valid)
            act3[(size_t)b * 3136 + (p0 + m) * 64 + (n0 + nn)] = (bf16)relu(acc[r] + bias);
    }
}

// ---- Kernel 4: per-expert fused 6-layer MLP + gather ---------------------
// grid = 8 experts * (B/16), block = 128. Layer-1 A tiles stream via TDM
// (tensor_load_to_lds), double-buffered against WMMA.
__global__ __launch_bounds__(128) void mlp_kernel(
    const bf16* __restrict__ flat, const int* __restrict__ rm,
    const bf16* __restrict__ W1p, const float* __restrict__ B1,
    const bf16* __restrict__ W2p, const float* __restrict__ B2,
    const bf16* __restrict__ W3p, const float* __restrict__ B3,
    const bf16* __restrict__ W4p, const float* __restrict__ B4,
    const bf16* __restrict__ W5p, const float* __restrict__ B5,
    const float* __restrict__ W6, const float* __restrict__ B6,
    float* __restrict__ out, int tilesPerE)
{
    const int CH = 224, NCH = 14;  // 14 * 224 = 3136 exactly
    __shared__ __align__(16) bf16 aC[2][16 * CH];
    __shared__ __align__(16) bf16 hA[16 * 64];
    __shared__ __align__(16) bf16 hB[16 * 64];
    __shared__ int rmL[16];
    const int e = blockIdx.x / tilesPerE;
    const int t0 = (blockIdx.x % tilesPerE) * 16;
    const int tid = threadIdx.x;
    if (tid < 16) rmL[tid] = rm[t0 + tid];
    __syncthreads();
    bool any = false;
    #pragma unroll
    for (int i = 0; i < 16; ++i) any |= (rmL[i] == e);
    if (!any) return;  // uniform across block

    const int lane = tid & 31, wv = tid >> 5;
    const int half = lane >> 4, nn = lane & 15, n0 = wv * 16;

    // ---- layer 1: [16,3136] x [3136,64], A via TDM double buffer ----
    v8f acc = {0.f, 0.f, 0.f, 0.f, 0.f, 0.f, 0.f, 0.f};
    const bf16* w1row = W1p + ((size_t)e * 64 + n0 + nn) * 3136;
    const bf16* flatBase = flat + (size_t)t0 * 3136;
    const unsigned int lds0 = (unsigned int)(size_t)(void*)&aC[0][0];
    const unsigned int lds1 = (unsigned int)(size_t)(void*)&aC[1][0];
    if (wv == 0)
        tdm_load_2d(lds0, flatBase, CH, 16, 3136, 3136, 16);
    for (int ic = 0; ic < NCH; ++ic) {
        if (wv == 0) {
            if (ic + 1 < NCH) {
                tdm_load_2d((ic & 1) ? lds0 : lds1, flatBase + (ic + 1) * CH,
                            CH, 16, 3136, 3136, 16);
                __builtin_amdgcn_s_wait_tensorcnt(1);  // chunk ic has landed
            } else {
                __builtin_amdgcn_s_wait_tensorcnt(0);
            }
        }
        __syncthreads();
        const bf16* aBuf = &aC[ic & 1][0];
        __builtin_prefetch(w1row + (ic + 1) * CH, 0, 0);
        #pragma unroll
        for (int kk = 0; kk < CH; kk += 32)
            acc = wmma_bf16(frag_a(aBuf + nn * CH + kk, half),
                            frag_b(w1row + ic * CH + kk, half), acc);
        __syncthreads();  // all waves done reading before TDM overwrites
    }
    {
        const float bias = B1[e * 64 + n0 + nn];
        #pragma unroll
        for (int r = 0; r < 8; ++r)
            hA[(r + 8 * half) * 64 + (n0 + nn)] = (bf16)relu(acc[r] + bias);
    }
    __syncthreads();

    // ---- layers 2..5: [16,64] x [64,64] ----
    const bf16* Wl[4] = {W2p, W3p, W4p, W5p};
    const float* Bl[4] = {B2, B3, B4, B5};
    bf16* hin = hA; bf16* hout = hB;
    #pragma unroll
    for (int L = 0; L < 4; ++L) {
        v8f a2 = {0.f, 0.f, 0.f, 0.f, 0.f, 0.f, 0.f, 0.f};
        const bf16* wrow = Wl[L] + ((size_t)e * 64 + n0 + nn) * 64;
        #pragma unroll
        for (int kk = 0; kk < 64; kk += 32)
            a2 = wmma_bf16(frag_a(hin + nn * 64 + kk, half),
                           frag_b(wrow + kk, half), a2);
        const float bias = Bl[L][e * 64 + n0 + nn];
        #pragma unroll
        for (int r = 0; r < 8; ++r)
            hout[(r + 8 * half) * 64 + (n0 + nn)] = (bf16)relu(a2[r] + bias);
        __syncthreads();
        bf16* t = hin; hin = hout; hout = t;
    }

    // ---- final layer: [16,64] x [64,6] + per-sample expert gather ----
    if (tid < 96) {
        int m = tid / 6, a = tid % 6;
        if (rmL[m] == e) {
            float s = B6[e * 6 + a];
            const float* w6 = W6 + (size_t)e * 64 * 6 + a;
            #pragma unroll
            for (int k = 0; k < 64; ++k) s += (float)hin[m * 64 + k] * w6[k * 6];
            out[(size_t)(t0 + m) * 6 + a] = s;
        }
    }
}

// ---- Host-side launcher ---------------------------------------------------
extern "C" void kernel_launch(void* const* d_in, const int* in_sizes, int n_in,
                              void* d_out, int out_size, void* d_ws, size_t ws_size,
                              hipStream_t stream) {
    const float* state = (const float*)d_in[0];
    const int*   rm    = (const int*)d_in[1];
    const float* k1 = (const float*)d_in[2];
    const float* c1 = (const float*)d_in[3];
    const float* k2 = (const float*)d_in[4];
    const float* c2 = (const float*)d_in[5];
    const float* k3 = (const float*)d_in[6];
    const float* c3 = (const float*)d_in[7];
    const float* W1 = (const float*)d_in[8];
    const float* B1 = (const float*)d_in[9];
    const float* W2 = (const float*)d_in[10];
    const float* B2 = (const float*)d_in[11];
    const float* W3 = (const float*)d_in[12];
    const float* B3 = (const float*)d_in[13];
    const float* W4 = (const float*)d_in[14];
    const float* B4 = (const float*)d_in[15];
    const float* W5 = (const float*)d_in[16];
    const float* B5 = (const float*)d_in[17];
    const float* W6 = (const float*)d_in[18];
    const float* B6 = (const float*)d_in[19];

    const int B = in_sizes[1];

    char* ws = (char*)d_ws;
    size_t off = 0;
    auto alloc = [&](size_t bytes) {
        size_t o = off;
        off = (off + bytes + 255) & ~(size_t)255;
        return o;
    };
    bf16* w1p = (bf16*)(ws + alloc(32 * 192 * 2));
    bf16* w2p = (bf16*)(ws + alloc(64 * 512 * 2));
    bf16* w3p = (bf16*)(ws + alloc(64 * 576 * 2));
    bf16* W1p = (bf16*)(ws + alloc((size_t)8 * 64 * 3136 * 2));
    bf16* W2p = (bf16*)(ws + alloc(8 * 64 * 64 * 2));
    bf16* W3p = (bf16*)(ws + alloc(8 * 64 * 64 * 2));
    bf16* W4p = (bf16*)(ws + alloc(8 * 64 * 64 * 2));
    bf16* W5p = (bf16*)(ws + alloc(8 * 64 * 64 * 2));
    bf16* act1 = (bf16*)(ws + alloc((size_t)B * 400 * 32 * 2));
    bf16* act2 = (bf16*)(ws + alloc((size_t)B * 81 * 64 * 2));
    bf16* act3 = (bf16*)(ws + alloc((size_t)B * 3136 * 2));
    if (off > ws_size) return;

    const int repackTotal = 32 * 192 + 64 * 512 + 64 * 576 + 8 * 64 * 3136 + 4 * 8 * 64 * 64;
    repack_kernel<<<(repackTotal + 255) / 256, 256, 0, stream>>>(
        k1, k2, k3, W1, W2, W3, W4, W5, w1p, w2p, w3p, W1p, W2p, W3p, W4p, W5p);

    conv1_kernel<<<B * 25, 64, 0, stream>>>(state, w1p, c1, act1);
    conv2_kernel<<<B * 6, 128, 0, stream>>>(act1, w2p, c2, act2);
    conv3_kernel<<<B * 4, 128, 0, stream>>>(act2, w3p, c3, act3);

    const int tilesPerE = B / 16;
    mlp_kernel<<<8 * tilesPerE, 128, 0, stream>>>(
        act3, rm, W1p, B1, W2p, B2, W3p, B3, W4p, B4, W5p, B5, W6, B6,
        (float*)d_out, tilesPerE);
}
